// Attention_12077448037050
// MI455X (gfx1250) — matrix operand order repaired and checked by reading, not compiled, not run on hardware
//
#include <hip/hip_runtime.h>

typedef __bf16 bf16;
typedef __attribute__((ext_vector_type(16))) __bf16 v16bf;
typedef __attribute__((ext_vector_type(8)))  __bf16 v8bf;
typedef __attribute__((ext_vector_type(8)))  float  v8f;
typedef __attribute__((ext_vector_type(4)))  int    v4i;

#define EMBED 1024
#define NHEAD 16
#define HDIM  64
#define BATCH 4
#define SEQ   2048
#define MTOT  (BATCH * SEQ)   // 8192

// ---- gfx1250 async VMEM->LDS staging (guarded; falls back to b128 copy) ----
#if __has_builtin(__builtin_amdgcn_global_load_async_to_lds_b128) && \
    __has_builtin(__builtin_amdgcn_s_wait_asynccnt)
#define HAS_ASYNC_LDS 1
#else
#define HAS_ASYNC_LDS 0
#endif

#define AS_GLOBAL __attribute__((address_space(1)))
#define AS_LDS    __attribute__((address_space(3)))

// copy 16 bytes global -> LDS (per lane)
__device__ inline void cp16(const bf16* g, bf16* l) {
#if HAS_ASYNC_LDS
  __builtin_amdgcn_global_load_async_to_lds_b128(
      (AS_GLOBAL v4i*)(unsigned long long)g,
      (AS_LDS v4i*)(unsigned)(unsigned long long)l,
      0, 0);
#else
  *(uint4*)l = *(const uint4*)g;
#endif
}
// wait for this wave's outstanding async LDS copies (before the block barrier)
__device__ inline void cp_join() {
#if HAS_ASYNC_LDS
  __builtin_amdgcn_s_wait_asynccnt(0);
#endif
}

// ---- float -> bf16 helpers (round-to-nearest-even, bit level) ----
__device__ inline bf16 f2bf(float f) {
  union { float f; unsigned u; } c; c.f = f;
  unsigned r = c.u + 0x7FFFu + ((c.u >> 16) & 1u);
  unsigned short hs = (unsigned short)(r >> 16);
  bf16 out; __builtin_memcpy(&out, &hs, 2); return out;
}
__device__ inline unsigned pack_bf2(float a, float b) {
  union { float f; unsigned u; } ca, cb; ca.f = a; cb.f = b;
  unsigned ra = (ca.u + 0x7FFFu + ((ca.u >> 16) & 1u)) >> 16;
  unsigned rb = (cb.u + 0x7FFFu + ((cb.u >> 16) & 1u)) & 0xFFFF0000u;
  return (ra & 0xFFFFu) | rb;
}

// ---- WMMA fragment loaders (CDNA5 ISA 7.12.2, wave32), fully vectorized ----
// A (16x32 bf16 MxK), row-major tile: lane L<16 -> M=L, elems{0..7}=K0..7,{8..15}=K16..23
//                                     lane L>=16 -> M=L-16, K+8 / K+24. Two ds_load_b128.
__device__ inline v16bf load_a_frag(const bf16* p, int ld, int lane) {
  int m = lane & 15;
  int koff = (lane >> 4) << 3;
  const bf16* rp = p + m * ld + koff;
  v8bf lo = *(const v8bf*)rp;
  v8bf hi = *(const v8bf*)(rp + 16);
  return __builtin_shufflevector(lo, hi, 0,1,2,3,4,5,6,7,8,9,10,11,12,13,14,15);
}
// B (32x16 bf16 KxN) from N-major storage (B[k][n] at p[n*ld + k]):
// lane L<16 -> N=L, elems{0..7}=K0..7,{8..15}=K16..23 ; lane L>=16 -> N=L-16, K+8 / K+24.
__device__ inline v16bf load_b_frag_nm(const bf16* p, int ld, int lane) {
  int n = lane & 15;
  int koff = (lane >> 4) << 3;
  const bf16* cp = p + n * ld + koff;
  v8bf lo = *(const v8bf*)cp;
  v8bf hi = *(const v8bf*)(cp + 16);
  return __builtin_shufflevector(lo, hi, 0,1,2,3,4,5,6,7,8,9,10,11,12,13,14,15);
}

__device__ inline v8f wmma_bf16(v16bf a, v16bf b, v8f c) {
  return __builtin_amdgcn_wmma_f32_16x16x32_bf16(false, a, false, b, (short)0, c, false, false);
}

// =====================================================================
// Prep A: x [8192,1024] fp32 -> bf16 (single pass)
// =====================================================================
__global__ __launch_bounds__(256) void conv_x_kernel(
    const float* __restrict__ x, bf16* __restrict__ xB) {
  size_t i = ((size_t)blockIdx.x * 256 + threadIdx.x) * 8;
  const float4* src = (const float4*)(x + i);
  float4 f0 = src[0], f1 = src[1];
  *(uint4*)(xB + i) = make_uint4(pack_bf2(f0.x, f0.y), pack_bf2(f0.z, f0.w),
                                 pack_bf2(f1.x, f1.y), pack_bf2(f1.z, f1.w));
}

// =====================================================================
// Prep B: wo [1024,1024] fp32 -> bf16 same layout (already N-major for out-proj)
// =====================================================================
__global__ __launch_bounds__(256) void conv_wo_kernel(
    const float* __restrict__ wo, bf16* __restrict__ woB) {
  int i = (blockIdx.x * 256 + threadIdx.x) * 4;
  float4 f = *(const float4*)(wo + i);
  uint2 u = make_uint2(pack_bf2(f.x, f.y), pack_bf2(f.z, f.w));
  *(uint2*)(woB + i) = u;
}

// =====================================================================
// Prep C: w[h][d][e] fp32 -> WT[h][e][d] bf16  (N-major weights), z = q/k/v
// =====================================================================
__global__ __launch_bounds__(256) void transpose_w_kernel(
    const float* __restrict__ wq, const float* __restrict__ wk,
    const float* __restrict__ wv,
    bf16* __restrict__ wqT, bf16* __restrict__ wkT, bf16* __restrict__ wvT) {
  int z = blockIdx.y;
  const float* w = (z == 0) ? wq : (z == 1) ? wk : wv;
  bf16* wt = (z == 0) ? wqT : (z == 1) ? wkT : wvT;
  int idx = (blockIdx.x * 256 + threadIdx.x) * 4;   // [h][e][d0..d0+3]
  int d0 = idx & (EMBED - 1);
  int he = idx >> 10;
  int h = he >> 6, e = he & 63;
  const float* src = w + ((size_t)h * EMBED + d0) * HDIM + e;
  uint2 u = make_uint2(pack_bf2(src[0], src[HDIM]),
                       pack_bf2(src[2 * HDIM], src[3 * HDIM]));
  *(uint2*)(wt + idx) = u;
}

// =====================================================================
// Kernel 1: QKV projection. grid = (M/128, H, 3), 256 thr (8 waves).
// 128x64 C-tile per block, each wave owns 32x32 (4 WMMA / k-step).
// Staging via async VMEM->LDS (ASYNCcnt) when available.
// Q,K stored [b,h,s,e] bf16 ; V stored transposed [b,h,e,s] bf16.
// Q pre-scaled by 1/sqrt(HDIM).
// =====================================================================
__global__ __launch_bounds__(256) void qkv_proj_kernel(
    const bf16* __restrict__ xB,
    const bf16* __restrict__ wqT, const bf16* __restrict__ wkT,
    const bf16* __restrict__ wvT,
    const float* __restrict__ bq, const float* __restrict__ bk,
    const float* __restrict__ bv,
    bf16* __restrict__ Q, bf16* __restrict__ K, bf16* __restrict__ VT) {
  const int mTile = blockIdx.x;
  const int h = blockIdx.y;
  const int z = blockIdx.z;                    // 0=Q 1=K 2=V
  const bf16*  wt   = (z == 0) ? wqT : (z == 1) ? wkT : wvT;
  const float* bias = (z == 0) ? bq : (z == 1) ? bk : bv;

  const int tid = threadIdx.x;
  const int lane = tid & 31;
  const int wid = tid >> 5;
  const int mBase = (wid >> 1) * 32;
  const int nBase = (wid & 1) * 32;

  __shared__ alignas(16) bf16 sA[128][40];   // A: 128x32 row-major (+pad)
  __shared__ alignas(16) bf16 sB[64][40];    // B: N-major [n][k], 64x32

  v8f acc00 = {}, acc01 = {}, acc10 = {}, acc11 = {};
  const bf16* wth = wt + (size_t)h * HDIM * EMBED;   // [e][d]
  const int m0 = mTile * 128;

  for (int k0 = 0; k0 < EMBED; k0 += 32) {
    {   // stage A: xB[m0..+127][k0..+31], 32 bytes / thread
      int r = tid >> 1, c = (tid & 1) * 16;
      const bf16* src = xB + (size_t)(m0 + r) * EMBED + k0 + c;
      cp16(src, &sA[r][c]);
      cp16(src + 8, &sA[r][c + 8]);
    }
    {   // stage B (N-major): WT[h][n][k0..+31], 16 bytes / thread
      int n = tid >> 2, kc = (tid & 3) * 8;
      cp16(wth + (size_t)n * EMBED + k0 + kc, &sB[n][kc]);
    }
    if (k0 + 32 < EMBED)
      __builtin_prefetch(xB + (size_t)(m0 + (tid >> 1)) * EMBED + k0 + 32, 0, 1);
    cp_join();
    __syncthreads();
    v16bf a0 = load_a_frag(&sA[mBase][0], 40, lane);
    v16bf a1 = load_a_frag(&sA[mBase + 16][0], 40, lane);
    v16bf b0 = load_b_frag_nm(&sB[nBase][0], 40, lane);
    v16bf b1 = load_b_frag_nm(&sB[nBase + 16][0], 40, lane);
    acc00 = wmma_bf16(a0, b0, acc00);
    acc01 = wmma_bf16(a0, b1, acc01);
    acc10 = wmma_bf16(a1, b0, acc10);
    acc11 = wmma_bf16(a1, b1, acc11);
    __syncthreads();
  }

  const int b = m0 >> 11;
  const int sRow0 = m0 & (SEQ - 1);
  const int col = nBase + (lane & 15);
  const int rowOff = ((lane >> 4) << 3);
  const float qscale = (z == 0) ? 0.125f : 1.0f;
  const float bias0 = bias[h * HDIM + col];
  const float bias1 = bias[h * HDIM + col + 16];
#pragma unroll
  for (int g = 0; g < 2; ++g) {          // two 16-row groups (acc0x / acc1x)
    v8f c0 = g ? acc10 : acc00;
    v8f c1 = g ? acc11 : acc01;
    int rowG = mBase + g * 16 + rowOff;
#pragma unroll
    for (int r = 0; r < 8; ++r) {
      int srow = sRow0 + rowG + r;
      float v0 = (c0[r] + bias0) * qscale;
      float v1 = (c1[r] + bias1) * qscale;
      size_t baseSE = ((size_t)(b * NHEAD + h) * SEQ + srow) * HDIM;
      if (z == 0) {
        Q[baseSE + col] = f2bf(v0); Q[baseSE + col + 16] = f2bf(v1);
      } else if (z == 1) {
        K[baseSE + col] = f2bf(v0); K[baseSE + col + 16] = f2bf(v1);
      } else {
        size_t baseE = (size_t)(b * NHEAD + h) * HDIM;
        VT[(baseE + col) * SEQ + srow] = f2bf(v0);
        VT[(baseE + col + 16) * SEQ + srow] = f2bf(v1);
      }
    }
  }
}

// =====================================================================
// Kernel 2: fused flash attention. grid = (S/64, H, B), 256 thr.
// Scores stay in LDS; online softmax; O accum in registers.
// att written bf16 as [b, s, h, e]  (== [8192, 1024] row-major).
// =====================================================================
#define LDT 72
#define LDSS 68

__global__ __launch_bounds__(256) void flash_attn_kernel(
    const bf16* __restrict__ Q, const bf16* __restrict__ K,
    const bf16* __restrict__ VT, bf16* __restrict__ att) {
  const int qTile = blockIdx.x;
  const int h = blockIdx.y;
  const int b = blockIdx.z;

  const int tid = threadIdx.x;
  const int lane = tid & 31;
  const int wid = tid >> 5;
  const int mBase = (wid >> 1) * 16;
  const int nBase = (wid & 1) * 32;

  __shared__ alignas(16) bf16 sQ[64][LDT];    // [s][e] row-major (A operand)
  __shared__ alignas(16) bf16 sK[64][LDT];    // [t][e]  == N-major B for scores
  __shared__ alignas(16) bf16 sVT[64][LDT];   // [e][t]  == N-major B for P*V
  __shared__ alignas(16) bf16 sP[64][LDT];    // [s][t] row-major (A operand)
  __shared__ alignas(16) float sS[64][LDSS];
  __shared__ float sM[64], sL[64], sF[64];
  __shared__ float sPart[64][4];

  const size_t bh = (size_t)(b * NHEAD + h);
  const bf16* Qp  = Q  + bh * SEQ * HDIM;
  const bf16* Kp  = K  + bh * SEQ * HDIM;
  const bf16* VTp = VT + bh * HDIM * SEQ;

  {   // stage Q tile (64x64) once: 32 bytes / thread
    int r = tid >> 2, c = (tid & 3) * 16;
    const bf16* src = Qp + (size_t)(qTile * 64 + r) * HDIM + c;
    cp16(src, &sQ[r][c]);
    cp16(src + 8, &sQ[r][c + 8]);
  }
  if (tid < 64) { sM[tid] = -3.0e38f; sL[tid] = 0.0f; }
  cp_join();
  __syncthreads();

  v16bf aq0 = load_a_frag(&sQ[mBase][0],  LDT, lane);
  v16bf aq1 = load_a_frag(&sQ[mBase][32], LDT, lane);
  v8f o0 = {}; v8f o1 = {};

  for (int j = 0; j < SEQ / 64; ++j) {
    {   // stage K tile [t][e]
      int r = tid >> 2, c = (tid & 3) * 16;
      const bf16* src = Kp + (size_t)(j * 64 + r) * HDIM + c;
      cp16(src, &sK[r][c]);
      cp16(src + 8, &sK[r][c + 8]);
    }
    {   // stage V^T tile [e][t]
      int r = tid >> 2, c = (tid & 3) * 16;
      const bf16* src = VTp + (size_t)r * SEQ + j * 64 + c;
      cp16(src, &sVT[r][c]);
      cp16(src + 8, &sVT[r][c + 8]);
    }
    cp_join();
    __syncthreads();

    // S = Q*K^T (scale folded into Q).  B[k=e][n=t] N-major == sK[t][e].
    v8f s0 = {}; v8f s1 = {};
    {
      v16bf bb0 = load_b_frag_nm(&sK[nBase][0],       LDT, lane);
      v16bf bb1 = load_b_frag_nm(&sK[nBase + 16][0],  LDT, lane);
      s0 = wmma_bf16(aq0, bb0, s0);
      s1 = wmma_bf16(aq0, bb1, s1);
      bb0 = load_b_frag_nm(&sK[nBase][32],      LDT, lane);
      bb1 = load_b_frag_nm(&sK[nBase + 16][32], LDT, lane);
      s0 = wmma_bf16(aq1, bb0, s0);
      s1 = wmma_bf16(aq1, bb1, s1);
    }
    {   // spill scores to LDS (C layout: VGPR r -> row r / r+8)
      int row = mBase + ((lane >> 4) << 3);
      int col = nBase + (lane & 15);
#pragma unroll
      for (int r = 0; r < 8; ++r) {
        sS[row + r][col] = s0[r];
        sS[row + r][col + 16] = s1[r];
      }
    }
    __syncthreads();

    {   // partial row max (4 threads per row)
      int r = tid >> 2, c = (tid & 3) * 16;
      float pm = sS[r][c];
#pragma unroll
      for (int i = 1; i < 16; ++i) pm = fmaxf(pm, sS[r][c + i]);
      sPart[r][tid & 3] = pm;
    }
    __syncthreads();
    if (tid < 64) {   // online softmax state update
      float rm = fmaxf(fmaxf(sPart[tid][0], sPart[tid][1]),
                       fmaxf(sPart[tid][2], sPart[tid][3]));
      float mOld = sM[tid];
      float mNew = fmaxf(mOld, rm);
      float f = __expf(mOld - mNew);
      sM[tid] = mNew; sF[tid] = f; sL[tid] *= f;
    }
    __syncthreads();

    {   // rescale O accumulators by per-row factor
      int row = mBase + ((lane >> 4) << 3);
#pragma unroll
      for (int r = 0; r < 8; ++r) {
        float f = sF[row + r];
        o0[r] *= f; o1[r] *= f;
      }
    }
    {   // P = exp(S - m), packed bf16 stores, + partial row sums
      int r = tid >> 2, c = (tid & 3) * 16;
      float mNew = sM[r];
      float psum = 0.0f;
      unsigned up[8];
#pragma unroll
      for (int i = 0; i < 8; ++i) {
        float p0 = __expf(sS[r][c + 2 * i] - mNew);
        float p1 = __expf(sS[r][c + 2 * i + 1] - mNew);
        psum += p0 + p1;
        up[i] = pack_bf2(p0, p1);
      }
      *(uint4*)&sP[r][c]     = make_uint4(up[0], up[1], up[2], up[3]);
      *(uint4*)&sP[r][c + 8] = make_uint4(up[4], up[5], up[6], up[7]);
      sPart[r][tid & 3] = psum;
    }
    __syncthreads();
    if (tid < 64)
      sL[tid] += sPart[tid][0] + sPart[tid][1] + sPart[tid][2] + sPart[tid][3];

    {   // O += P * V.  B[k=t][n=e] N-major == sVT[e][t].
      v16bf ap0 = load_a_frag(&sP[mBase][0],  LDT, lane);
      v16bf ap1 = load_a_frag(&sP[mBase][32], LDT, lane);
      v16bf bb0 = load_b_frag_nm(&sVT[nBase][0],      LDT, lane);
      v16bf bb1 = load_b_frag_nm(&sVT[nBase + 16][0], LDT, lane);
      o0 = wmma_bf16(ap0, bb0, o0);
      o1 = wmma_bf16(ap0, bb1, o1);
      bb0 = load_b_frag_nm(&sVT[nBase][32],      LDT, lane);
      bb1 = load_b_frag_nm(&sVT[nBase + 16][32], LDT, lane);
      o0 = wmma_bf16(ap1, bb0, o0);
      o1 = wmma_bf16(ap1, bb1, o1);
    }
    __syncthreads();
  }

  {   // normalize + store att[b][s][h][e]
    int row = mBase + ((lane >> 4) << 3);
    int col = nBase + (lane & 15);
#pragma unroll
    for (int r = 0; r < 8; ++r) {
      int s = qTile * 64 + row + r;
      float rl = 1.0f / sL[row + r];
      size_t base = (((size_t)b * SEQ + s) * NHEAD + h) * HDIM;
      att[base + col] = f2bf(o0[r] * rl);
      att[base + col + 16] = f2bf(o1[r] * rl);
    }
  }
}

// =====================================================================
// Kernel 3: out = att @ wo^T + bo.  grid = (M/128, E/64), 256 thr.
// woB (bf16, [e_out][e_in]) is already N-major for this GEMM.
// =====================================================================
__global__ __launch_bounds__(256) void out_proj_kernel(
    const bf16* __restrict__ att, const bf16* __restrict__ woB,
    const float* __restrict__ bo, float* __restrict__ out) {
  const int mTile = blockIdx.x;
  const int nTile = blockIdx.y;
  const int tid = threadIdx.x;
  const int lane = tid & 31;
  const int wid = tid >> 5;
  const int mBase = (wid >> 1) * 32;
  const int nBase = (wid & 1) * 32;

  __shared__ alignas(16) bf16 sA[128][40];
  __shared__ alignas(16) bf16 sB[64][40];   // N-major [n][k]

  v8f acc00 = {}, acc01 = {}, acc10 = {}, acc11 = {};
  const int m0 = mTile * 128;
  const int n0 = nTile * 64;

  for (int k0 = 0; k0 < EMBED; k0 += 32) {
    {   // stage A: att bf16, 32 bytes / thread
      int r = tid >> 1, c = (tid & 1) * 16;
      const bf16* src = att + (size_t)(m0 + r) * EMBED + k0 + c;
      cp16(src, &sA[r][c]);
      cp16(src + 8, &sA[r][c + 8]);
    }
    {   // stage B: woB[n0+n][k0..+31]
      int n = tid >> 2, kc = (tid & 3) * 8;
      cp16(woB + (size_t)(n0 + n) * EMBED + k0 + kc, &sB[n][kc]);
    }
    cp_join();
    __syncthreads();
    v16bf a0 = load_a_frag(&sA[mBase][0], 40, lane);
    v16bf a1 = load_a_frag(&sA[mBase + 16][0], 40, lane);
    v16bf b0 = load_b_frag_nm(&sB[nBase][0], 40, lane);
    v16bf b1 = load_b_frag_nm(&sB[nBase + 16][0], 40, lane);
    acc00 = wmma_bf16(a0, b0, acc00);
    acc01 = wmma_bf16(a0, b1, acc01);
    acc10 = wmma_bf16(a1, b0, acc10);
    acc11 = wmma_bf16(a1, b1, acc11);
    __syncthreads();
  }

  {
    int rowOff = ((lane >> 4) << 3);
    int col = nBase + (lane & 15);
    float bo0 = bo[n0 + col], bo1 = bo[n0 + col + 16];
#pragma unroll
    for (int g = 0; g < 2; ++g) {
      v8f c0 = g ? acc10 : acc00;
      v8f c1 = g ? acc11 : acc01;
      int rowG = mBase + g * 16 + rowOff;
#pragma unroll
      for (int r = 0; r < 8; ++r) {
        size_t o = (size_t)(m0 + rowG + r) * EMBED + n0;
        out[o + col] = c0[r] + bo0;
        out[o + col + 16] = c1[r] + bo1;
      }
    }
  }
}

// =====================================================================
extern "C" void kernel_launch(void* const* d_in, const int* in_sizes, int n_in,
                              void* d_out, int out_size, void* d_ws, size_t ws_size,
                              hipStream_t stream) {
  (void)in_sizes; (void)n_in; (void)out_size; (void)ws_size;
  const float* x  = (const float*)d_in[0];
  const float* wq = (const float*)d_in[1];
  const float* bq = (const float*)d_in[2];
  const float* wk = (const float*)d_in[3];
  const float* bk = (const float*)d_in[4];
  const float* wv = (const float*)d_in[5];
  const float* bv = (const float*)d_in[6];
  const float* wo = (const float*)d_in[7];
  const float* bo = (const float*)d_in[8];
  float* out = (float*)d_out;

  bf16* ws = (bf16*)d_ws;
  const size_t nQKV = (size_t)BATCH * NHEAD * SEQ * HDIM;   // 8,388,608
  const size_t nW   = (size_t)EMBED * EMBED;                // 1,048,576
  const size_t nX   = (size_t)MTOT * EMBED;                 // 8,388,608
  bf16* Q   = ws;
  bf16* K   = Q   + nQKV;
  bf16* VT  = K   + nQKV;
  bf16* att = VT  + nQKV;
  bf16* xB  = att + nQKV;
  bf16* woB = xB  + nX;
  bf16* wqT = woB + nW;
  bf16* wkT = wqT + nW;
  bf16* wvT = wkT + nW;

  hipLaunchKernelGGL(conv_x_kernel, dim3(nX / 2048), dim3(256), 0, stream, x, xB);
  hipLaunchKernelGGL(conv_wo_kernel, dim3(nW / 1024), dim3(256), 0, stream, wo, woB);
  hipLaunchKernelGGL(transpose_w_kernel, dim3(nW / 1024, 3), dim3(256), 0, stream,
                     wq, wk, wv, wqT, wkT, wvT);
  hipLaunchKernelGGL(qkv_proj_kernel, dim3(MTOT / 128, NHEAD, 3), dim3(256), 0, stream,
                     xB, wqT, wkT, wvT, bq, bk, bv, Q, K, VT);
  hipLaunchKernelGGL(flash_attn_kernel, dim3(SEQ / 64, NHEAD, BATCH), dim3(256), 0, stream,
                     Q, K, VT, att);
  hipLaunchKernelGGL(out_proj_kernel, dim3(MTOT / 128, NHEAD), dim3(256), 0, stream,
                     att, woB, bo, out);
}
